// CV_STDPModel_10823317586345
// MI455X (gfx1250) — matrix operand order repaired and checked by reading, not compiled, not run on hardware
//
#include <hip/hip_runtime.h>
#include <hip/hip_bf16.h>

typedef __attribute__((ext_vector_type(16))) _Float16 v16h;
typedef __attribute__((ext_vector_type(8)))  float    v8f;

#define T_STEPS 30

// ---------------------------------------------------------------------------
// K0: zero the global-max accumulator word
// ---------------------------------------------------------------------------
__global__ void zero_max_kernel(unsigned* m) {
    if (threadIdx.x == 0 && blockIdx.x == 0) *m = 0u;
}

// ---------------------------------------------------------------------------
// K1: DOG preprocess. One block per image.
//   b = gauss_blur over (batch, channel[identity], H, W), d = |b - x|
//   writes d to scratch, atomicMax of d into global max (nonneg -> uint bits).
// ---------------------------------------------------------------------------
__global__ __launch_bounds__(256)
void dog_kernel(const float* __restrict__ x, int N,
                float* __restrict__ d_dog, unsigned* __restrict__ maxbits) {
    __shared__ float bb[784];   // batch-blurred
    __shared__ float xx[784];   // original image
    __shared__ float tb[784];   // row-blurred
    __shared__ float red[256];

    const int n = blockIdx.x;
    const int tid = threadIdx.x;
    // normalized 7-tap gaussian, sigma=2, radius=3
    const float G[7] = {0.0701593186f, 0.1310738663f, 0.1907164954f,
                        0.2161006392f,
                        0.1907164954f, 0.1310738663f, 0.0701593186f};

    // blur along batch axis (symmetric padding), center tap is x[n] itself
    for (int i = tid; i < 784; i += 256) {
        float acc = 0.f;
        for (int k = 0; k < 7; ++k) {
            int nn = n + k - 3;
            nn = (nn < 0) ? (-1 - nn) : nn;
            nn = (nn >= N) ? (2 * N - 1 - nn) : nn;
            float xv = x[(size_t)nn * 784 + i];
            if (k == 3) xx[i] = xv;
            acc += G[k] * xv;
        }
        bb[i] = acc;   // channel axis (size 1, symmetric pad) is identity
    }
    __syncthreads();

    // blur along H (rows of 28), symmetric padding
    for (int i = tid; i < 784; i += 256) {
        int r = i / 28, c = i - r * 28;
        float acc = 0.f;
        for (int k = 0; k < 7; ++k) {
            int rr = r + k - 3;
            rr = (rr < 0) ? (-1 - rr) : rr;
            rr = (rr >= 28) ? (55 - rr) : rr;
            acc += G[k] * bb[rr * 28 + c];
        }
        tb[i] = acc;
    }
    __syncthreads();

    // blur along W, then d = |b - x|, write out + track max
    float lmax = 0.f;
    for (int i = tid; i < 784; i += 256) {
        int r = i / 28, c = i - r * 28;
        float acc = 0.f;
        for (int k = 0; k < 7; ++k) {
            int cc = c + k - 3;
            cc = (cc < 0) ? (-1 - cc) : cc;
            cc = (cc >= 28) ? (55 - cc) : cc;
            acc += G[k] * tb[r * 28 + cc];
        }
        float d = fabsf(acc - xx[i]);
        d_dog[(size_t)n * 784 + i] = d;
        lmax = fmaxf(lmax, d);
    }
    red[tid] = lmax;
    __syncthreads();
    for (int s = 128; s > 0; s >>= 1) {
        if (tid < s) red[tid] = fmaxf(red[tid], red[tid + s]);
        __syncthreads();
    }
    if (tid == 0) atomicMax(maxbits, __float_as_uint(red[0]));
}

// ---------------------------------------------------------------------------
// K2: 30-step SNN, one block (8 wave32) per image.
//   - Membrane potentials in registers (fixed lane ownership over timesteps).
//   - Spike maps in LDS (cross-thread for pooling / im2col).
//   - conv1/conv2 via v_wmma_f32_16x16x32_f16; WMMAs issued in independent
//     pairs so the 4-coexec WMMA->VALU hazard window is filled with useful
//     work instead of v_nops. IF neuron fused into branch-free epilogues
//     (bn>=4 lanes store to a per-lane dump region -> no saveexec).
// ---------------------------------------------------------------------------
#define S2_STRIDE 688              // 43 tiles * 16 px (>= 676), padded
#define S2_SIZE   (4 * S2_STRIDE + 256)   // + per-lane dump region
#define S3_STRIDE 128              // 8 tiles * 16 px (>= 121)
#define S3_SIZE   (4 * S3_STRIDE + 256)

__global__ __launch_bounds__(256)
void snn_kernel(const float* __restrict__ d_dog,
                const unsigned* __restrict__ maxbits,
                const float* __restrict__ w1,   // [4][1][3][3]
                const float* __restrict__ w2,   // [4][4][3][3]
                const float* __restrict__ w3,   // [1][4][3][3]
                float* __restrict__ out) {
    __shared__ _Float16 s1h[784];        // node1 spikes, 28x28
    __shared__ _Float16 s2ex[S2_SIZE];   // node2 spikes [4][688] + dump
    __shared__ _Float16 p2s[676];        // pooled, 4 x 13 x 13
    __shared__ _Float16 s3ex[S3_SIZE];   // node3 spikes [4][128] + dump
    __shared__ _Float16 p3s[144];        // pooled (pad=1), 4 x 6 x 6
    __shared__ float    w3s[36];

    const int n    = blockIdx.x;
    const int tid  = threadIdx.x;
    const int lane = tid & 31;
    const int wave = tid >> 5;
    const int bn   = lane & 15;   // B/D column (out channel) & A row-in-tile
    const int hi   = lane >> 4;   // lane half: selects K sub-range

    const float invmax = 1.0f / __uint_as_float(*maxbits);

    // ---- register-resident state ----
    float dnr[4], v1r[4];                 // node1 input & membrane (4 px/thread)
#pragma unroll
    for (int s = 0; s < 4; ++s) {
        int i = tid + s * 256;
        dnr[s] = (i < 784) ? d_dog[(size_t)n * 784 + i] * invmax : 0.f;
        v1r[s] = 0.f;
    }
    float v2r[6][8];                      // node2 membrane (own conv1 outputs)
#pragma unroll
    for (int s = 0; s < 6; ++s)
#pragma unroll
        for (int r = 0; r < 8; ++r) v2r[s][r] = 0.f;
    float v3r[8];                         // node3 membrane (own conv2 outputs)
#pragma unroll
    for (int r = 0; r < 8; ++r) v3r[r] = 0.f;
    float v4r = 0.f, accr = 0.f;          // node4 (LIF) membrane & spike count

    if (tid < 36) w3s[tid] = w3[tid];

    // --- constant B fragments (built once, kept in VGPRs across T loop) ---
    // B 32x16 layout (matches A): elem j -> K=8*hi+j (j<8), K=16+8*hi+j-8 (j>=8);
    // column n=lane%16.
    v16h bw1, bw2a, bw2b;
#pragma unroll
    for (int j = 0; j < 16; ++j) {
        int k = hi * 8 + j + (j >= 8 ? 8 : 0);
        bw1[j]  = (_Float16)((bn < 4 && k < 9) ? w1[bn * 9 + k] : 0.f);       // K = 3x3 taps
        bw2a[j] = (_Float16)((bn < 4)          ? w2[bn * 36 + k] : 0.f);      // k 0..31 (<36)
        bw2b[j] = (_Float16)((bn < 4 && k < 4) ? w2[bn * 36 + 32 + k] : 0.f); // k 32..35
    }
    __syncthreads();

    const int kbaseA = hi * 8;  // A layout: elem j -> K=kbaseA+j (j<8), K=16+kbaseA+j-8 (j>=8)
    const int dump2  = 4 * S2_STRIDE + lane * 8;   // per-lane dump base (no conflicts)
    const int dump3  = 4 * S3_STRIDE + lane * 8;

    for (int t = 0; t < T_STEPS; ++t) {
        // ---- IF node1 (th 4.5), register state, spikes -> LDS ----
#pragma unroll
        for (int s = 0; s < 4; ++s) {
            int i = tid + s * 256;
            if (i < 784) {
                float v = v1r[s] + dnr[s];
                bool  sp = v >= 4.5f;
                s1h[i] = (_Float16)(sp ? 1.0f : 0.0f);
                v1r[s] = sp ? 0.f : v;
            }
        }
        __syncthreads();

        // ---- conv1 (28->26, 1->4 ch): 43 WMMA tiles, processed in PAIRS so the
        //      two independent WMMAs + epilogues hide each other's hazards ----
#pragma unroll
        for (int s = 0; s < 6; s += 2) {
            const int tl0 = wave + s * 8;          // wave-uniform tile ids
            const int tl1 = wave + (s + 1) * 8;
            const bool do1 = (tl1 < 43);           // tl0 < 43 always (s<=4, wave<=7)

            v16h a0, a1;
            {   // gather A for tile 0
                int px = tl0 * 16 + bn;
                int pc = (px < 676) ? px : 675;
                int oy = pc / 26, ox = pc - oy * 26;
#pragma unroll
                for (int j = 0; j < 8; ++j) {
                    int k  = kbaseA + j;
                    int kk = (k < 9) ? k : 0;
                    int ky = kk / 3, kx = kk - ky * 3;
                    _Float16 val = s1h[(oy + ky) * 28 + ox + kx];
                    a0[j]     = (k < 9) ? val : (_Float16)0.f;
                    a0[j + 8] = (_Float16)0.f;
                }
            }
            {   // gather A for tile 1 (clamped when absent; result discarded)
                int tlc = do1 ? tl1 : tl0;
                int px = tlc * 16 + bn;
                int pc = (px < 676) ? px : 675;
                int oy = pc / 26, ox = pc - oy * 26;
#pragma unroll
                for (int j = 0; j < 8; ++j) {
                    int k  = kbaseA + j;
                    int kk = (k < 9) ? k : 0;
                    int ky = kk / 3, kx = kk - ky * 3;
                    _Float16 val = s1h[(oy + ky) * 28 + ox + kx];
                    a1[j]     = (k < 9) ? val : (_Float16)0.f;
                    a1[j + 8] = (_Float16)0.f;
                }
            }
            v8f c0 = {}, c1 = {};
            c0 = __builtin_amdgcn_wmma_f32_16x16x32_f16(false, a0, false, bw1,
                                                        (short)0, c0, false, false);
            c1 = __builtin_amdgcn_wmma_f32_16x16x32_f16(false, a1, false, bw1,
                                                        (short)0, c1, false, false);
            // epilogue tile 0 (branch-free; bn>=4 lanes -> dump region)
#pragma unroll
            for (int r = 0; r < 8; ++r) {
                int pix2 = tl0 * 16 + hi * 8 + r;  // D: vgpr r -> M=r+8*hi, N=bn
                float v  = v2r[s][r] + c0[r];
                bool  sp = v >= 4.5f;
                int  idx = (bn < 4) ? (bn * S2_STRIDE + pix2) : (dump2 + r);
                s2ex[idx]  = (_Float16)(sp ? 1.0f : 0.0f);
                v2r[s][r]  = sp ? 0.f : v;
            }
            // epilogue tile 1 (wave-uniform guard)
            if (do1) {
#pragma unroll
                for (int r = 0; r < 8; ++r) {
                    int pix2 = tl1 * 16 + hi * 8 + r;
                    float v  = v2r[s + 1][r] + c1[r];
                    bool  sp = v >= 4.5f;
                    int  idx = (bn < 4) ? (bn * S2_STRIDE + pix2) : (dump2 + r);
                    s2ex[idx]     = (_Float16)(sp ? 1.0f : 0.0f);
                    v2r[s + 1][r] = sp ? 0.f : v;
                }
            }
        }
        __syncthreads();

        // ---- maxpool 2x2, 26 -> 13 (spikes: max == OR) ----
        for (int i = tid; i < 676; i += 256) {
            int oc = i / 169, rem = i - oc * 169;
            int py = rem / 13, pxx = rem - py * 13;
            int b  = oc * S2_STRIDE + (2 * py) * 26 + 2 * pxx;
            float mm = fmaxf(fmaxf((float)s2ex[b],      (float)s2ex[b + 1]),
                             fmaxf((float)s2ex[b + 26], (float)s2ex[b + 27]));
            p2s[i] = (_Float16)mm;
        }
        __syncthreads();

        // ---- conv2 (13->11, 4->4 ch): 8 tiles, K=36 split into TWO INDEPENDENT
        //      WMMAs (both from C=0, summed in epilogue) -> no WMMA->WMMA RAW ----
        {
            int tl = wave;                          // exactly 8 tiles
            int px = tl * 16 + bn;
            int pc = (px < 121) ? px : 120;
            int oy = pc / 11, ox = pc - oy * 11;
            v16h a0, a1;
#pragma unroll
            for (int j = 0; j < 8; ++j) {
                {   int k = kbaseA + j;             // K 0..15
                    int ic = k / 9, r = k - ic * 9; int ky = r / 3, kx = r - ky * 3;
                    a0[j] = p2s[ic * 169 + (oy + ky) * 13 + ox + kx]; }
                {   int k = 16 + kbaseA + j;        // K 16..31
                    int ic = k / 9, r = k - ic * 9; int ky = r / 3, kx = r - ky * 3;
                    a0[j + 8] = p2s[ic * 169 + (oy + ky) * 13 + ox + kx]; }
                {   int k  = 32 + kbaseA + j;       // K' -> original k 32..47, valid < 36
                    int kk = (k < 36) ? k : 35;
                    int ic = kk / 9, r = kk - ic * 9; int ky = r / 3, kx = r - ky * 3;
                    _Float16 val = p2s[ic * 169 + (oy + ky) * 13 + ox + kx];
                    a1[j]     = (k < 36) ? val : (_Float16)0.f;
                    a1[j + 8] = (_Float16)0.f; }
            }
            v8f c0 = {}, c1 = {};
            c0 = __builtin_amdgcn_wmma_f32_16x16x32_f16(false, a0, false, bw2a,
                                                        (short)0, c0, false, false);
            c1 = __builtin_amdgcn_wmma_f32_16x16x32_f16(false, a1, false, bw2b,
                                                        (short)0, c1, false, false);
#pragma unroll
            for (int r = 0; r < 8; ++r) {
                int pix2 = tl * 16 + hi * 8 + r;
                float v  = v3r[r] + (c0[r] + c1[r]);
                bool  sp = v >= 1.0f;
                int  idx = (bn < 4) ? (bn * S3_STRIDE + pix2) : (dump3 + r);
                s3ex[idx] = (_Float16)(sp ? 1.0f : 0.0f);
                v3r[r]    = sp ? 0.f : v;
            }
        }
        __syncthreads();

        // ---- maxpool 2x2 pad=1 (-inf pad; spikes>=0), 11 -> 6 ----
        for (int i = tid; i < 144; i += 256) {
            int oc = i / 36, rem = i - oc * 36;
            int py = rem / 6, pxx = rem - py * 6;
            float mm = 0.f;
#pragma unroll
            for (int dy = 0; dy < 2; ++dy)
#pragma unroll
                for (int dx = 0; dx < 2; ++dx) {
                    int iy = 2 * py - 1 + dy, ix = 2 * pxx - 1 + dx;
                    if (iy >= 0 && iy < 11 && ix >= 0 && ix < 11)
                        mm = fmaxf(mm, (float)s3ex[oc * S3_STRIDE + iy * 11 + ix]);
                }
            p3s[i] = (_Float16)mm;
        }
        __syncthreads();

        // ---- conv3 (6->4, 4->1 ch, 16 outputs) + LIF node4 (tau=2), register state ----
        if (tid < 16) {
            int oy = tid >> 2, ox = tid & 3;
            float h = 0.f;
#pragma unroll
            for (int ic = 0; ic < 4; ++ic)
#pragma unroll
                for (int ky = 0; ky < 3; ++ky)
#pragma unroll
                    for (int kx = 0; kx < 3; ++kx)
                        h += w3s[ic * 9 + ky * 3 + kx] *
                             (float)p3s[ic * 36 + (oy + ky) * 6 + ox + kx];
            float vv = v4r + (h - v4r) * 0.5f;
            bool  sp = vv >= 1.0f;
            accr += sp ? 1.0f : 0.0f;
            v4r   = sp ? 0.f : vv;
        }
        // no barrier needed: next write to p3s is separated by >=1 barrier
    }

    if (tid < 16) out[(size_t)n * 16 + tid] = accr * (1.0f / (float)T_STEPS);
}

// ---------------------------------------------------------------------------
extern "C" void kernel_launch(void* const* d_in, const int* in_sizes, int n_in,
                              void* d_out, int out_size, void* d_ws, size_t ws_size,
                              hipStream_t stream) {
    const float* x  = (const float*)d_in[0];
    const float* w1 = (const float*)d_in[1];
    const float* w2 = (const float*)d_in[2];
    const float* w3 = (const float*)d_in[3];
    float* out = (float*)d_out;

    const int N = in_sizes[0] / 784;           // batch (4096)

    float*    ws_d   = (float*)d_ws;                                    // N*784 floats
    unsigned* ws_max = (unsigned*)((char*)d_ws + (size_t)N * 784 * 4);  // 1 word

    zero_max_kernel<<<1, 32, 0, stream>>>(ws_max);
    dog_kernel<<<N, 256, 0, stream>>>(x, N, ws_d, ws_max);
    snn_kernel<<<N, 256, 0, stream>>>(ws_d, ws_max, w1, w2, w3, out);
}
